// TextVAE_30897994727709
// MI455X (gfx1250) — matrix operand order repaired and checked by reading, not compiled, not run on hardware
//
#include <hip/hip_runtime.h>
#include <hip/hip_bf16.h>

// ---------------------------------------------------------------------------
// TextVAE on gfx1250 (MI455X): persistent batch-tiled kernel.
// Each wave32 owns a 16-row batch tile (WMMA M=16) and runs the whole
// encoder(64 steps) + decoder(64 steps, argmax feedback) recurrence with all
// weights resident in LDS as f16 in WMMA-B layout. All matmuls use
// v_wmma_f32_16x16x32_f16 with f32 accumulation. The f32 gate scratch is
// column-major so each lane's 8-row accumulator column stores as one v8f
// (2x ds_store_b128) instead of 8 scalar ds_store_b32.
// ---------------------------------------------------------------------------

typedef __attribute__((ext_vector_type(16))) _Float16 v16h;
typedef __attribute__((ext_vector_type(8)))  _Float16 v8h;
typedef __attribute__((ext_vector_type(8)))  float    v8f;

#define WAVE_SYNC() asm volatile("s_wait_dscnt 0x0" ::: "memory")

// Problem constants (from the reference)
#define BATCH 8192
#define TSTEPS 64
#define MAXLEN 64
#define VOCAB 128
#define HDIM 50
#define LDIM 20
#define SOS_TOK 1

#define NWAVES 4
#define ROWS_PER_WAVE 16
#define ROWS_PER_WG (NWAVES * ROWS_PER_WAVE)
#define GI_COLS 160   // f32 scratch: 160 columns x 16 rows, column-major

// column-major index into gi/gh scratch
#define CM(c, m) (((c) << 4) + (m))

// ---- LDS layout -----------------------------------------------------------
// Weight union (halves). Encoder view and decoder view overlap.
#define EMB_E 0        // embed_enc padded [128][64]              (8192)
#define WIHF  8192     // Wih_f  kt=2 nt=10                       (10240)
#define WHHF  18432    // Whh_f  kt=2 nt=10                       (10240)
#define WIHB  28672    // Wih_b  kt=2 nt=10                       (10240)
#define WH2P  38912    // Wh2p   kt=2 nt=3                        (3072)
#define WZ2H  41984    // Wz2h   kt=1 nt=4                        (2048)
// decoder view
#define ETAB  0        // silu(embed_dec) padded [128][64]        (8192)
#define WIH1  8192     // Wih1   kt=3 nt=10                       (15360)
#define WHH1  23552    // Whh1   kt=2 nt=10                       (10240)
#define WIH2  33792    // Wih2   kt=2 nt=10                       (10240)
#define WHH2  44032    // Whh2   kt=2 nt=10                       (10240)
#define WH2O  54272    // Wh2o   kt=3 nt=8                        (12288)
#define WPOOL_HALVES 66560

// Bias pool (floats, zero-padded to tile widths)
#define BIHF 0
#define BHHF 160
#define BIHB 320
#define BHHB 480
#define BH2P 640
#define BZ2H 688
#define BIH1 752
#define BHH1 912
#define BIH2 1072
#define BHH2 1232
#define BH2O 1392
#define BIAS_FLOATS 1520

struct WaveScratch {
  float    gi[GI_COLS * 16];  // 10 KB, column-major (also reused for logits)
  float    gh[GI_COLS * 16];  // 10 KB, column-major
  _Float16 hbuf[16 * 64];     // enc h / dec h1 (row-major, K padded to 64)
  _Float16 h2buf[16 * 64];    // dec h2
  _Float16 x1buf[16 * 96];    // [e | z | 0], K padded to 96
  _Float16 hz2buf[16 * 96];   // [h2 | z | 0]
  _Float16 zbuf[16 * 32];     // z, K padded to 32
  int      tok[16];
  float    logZ[16];
};

struct __align__(32) SharedMem {
  _Float16 wpool[WPOOL_HALVES];   // 133,120 B
  float    bias[BIAS_FLOATS];     //   6,080 B
  WaveScratch ws[NWAVES];         // 127,488 B
};                                // total ~266.7 KB < 320 KB/WGP

// ---- math helpers ---------------------------------------------------------
__device__ inline float fsig(float x)  { return 1.f / (1.f + __expf(-x)); }
__device__ inline float ftanh(float x) { float e = __expf(2.f * x); return 1.f - 2.f / (e + 1.f); }

// ---- cooperative weight conversion: f32 global -> f16 WMMA-B layout in LDS
__device__ inline void load_wB(const float* __restrict__ W, int OUT, int IN,
                               int KT, int NT, _Float16* dst, int tid) {
  int total = KT * NT * 512;
  for (int i = tid; i < total; i += NWAVES * 32) {
    int j = i & 15;
    int l = (i >> 4) & 31;
    int tile = i >> 9;
    int nt = tile % NT, kt = tile / NT;
    int o = nt * 16 + (l & 15);
    int k = kt * 32 + ((l & 16) ? 16 : 0) + j;
    float v = (o < OUT && k < IN) ? W[o * IN + k] : 0.f;
    dst[i] = (_Float16)v;
  }
}

__device__ inline void load_emb(const float* __restrict__ E, _Float16* dst,
                                bool silu, int tid) {
  for (int i = tid; i < 128 * 64; i += NWAVES * 32) {
    int r = i >> 6, k = i & 63;
    float v = (k < HDIM) ? E[r * HDIM + k] : 0.f;
    if (silu) v = v * fsig(v);
    dst[i] = (_Float16)v;
  }
}

__device__ inline void load_bias(const float* __restrict__ src, int n, int npad,
                                 float* dst, int tid) {
  for (int i = tid; i < npad; i += NWAVES * 32) dst[i] = (i < n) ? src[i] : 0.f;
}

// ---- wave-level GEMM: (16 x 32*KT) @ B-tiles -> column-major f32 LDS ------
// rowPtr: per-lane pointer to row M=(lane&15) of the A activation buffer.
template <int KT, int NT>
__device__ inline void wave_gemm(const _Float16* __restrict__ rowPtr,
                                 const _Float16* __restrict__ wlds,
                                 const float* __restrict__ bias,
                                 float* __restrict__ outLds, int lane) {
  const int khalf = lane >> 4;      // 0: lanes 0-15, 1: lanes 16-31
  const int n     = lane & 15;
  const int mbase = khalf << 3;     // this lane writes rows mbase..mbase+7
  v16h af[KT];
#pragma unroll
  for (int kt = 0; kt < KT; ++kt) {
    // 16-bit A layout: elems 0-15 -> K = khalf*16 + j (matches B packing)
    const _Float16* p = rowPtr + kt * 32 + khalf * 16;
    v8h lo = *(const v8h*)p;
    v8h hi = *(const v8h*)(p + 8);
    v16h a;
#pragma unroll
    for (int i = 0; i < 8; ++i) { a[i] = lo[i]; a[i + 8] = hi[i]; }
    af[kt] = a;
  }
#pragma unroll
  for (int nt = 0; nt < NT; ++nt) {
    v8f acc = {};
#pragma unroll
    for (int kt = 0; kt < KT; ++kt) {
      v16h b = *(const v16h*)(wlds + (((kt * NT + nt) << 5) + lane) * 16);
      acc = __builtin_amdgcn_wmma_f32_16x16x32_f16(false, af[kt], false, b,
                                                   (short)0, acc, false, false);
    }
    float bc = bias[nt * 16 + n];
#pragma unroll
    for (int r = 0; r < 8; ++r) acc[r] += bc;
    // column (nt*16+n), rows mbase..mbase+7: contiguous 32B -> 2x ds_store_b128
    *(v8f*)(outLds + CM(nt * 16 + n, mbase)) = acc;
  }
}

// ---- GRU gate combine: h' = (1-z)*n + z*h ---------------------------------
// Full-wave mapping: m = lane&15, c = (lane>>4) + 2*i, i in [0,25)  -> 16x50.
__device__ inline void gru_combine(WaveScratch& W, _Float16* hdst,
                                   _Float16* aux, int lane) {
  const int m  = lane & 15;
  const int ch = lane >> 4;
  for (int i = 0; i < 25; ++i) {
    int c = ch + 2 * i;
    float r  = fsig(W.gi[CM(c, m)] + W.gh[CM(c, m)]);
    float zg = fsig(W.gi[CM(50 + c, m)] + W.gh[CM(50 + c, m)]);
    float nn = ftanh(W.gi[CM(100 + c, m)] + r * W.gh[CM(100 + c, m)]);
    float ho = (float)hdst[m * 64 + c];
    _Float16 hn = (_Float16)((1.f - zg) * nn + zg * ho);
    hdst[m * 64 + c] = hn;
    if (aux) aux[m * 96 + c] = hn;
  }
}

// ---------------------------------------------------------------------------
__global__ __launch_bounds__(NWAVES * 32, 1) void textvae_kernel(
    const int* __restrict__ x,
    const float* __restrict__ embed_enc, const float* __restrict__ Wih_f,
    const float* __restrict__ Whh_f, const float* __restrict__ bih_f,
    const float* __restrict__ bhh_f, const float* __restrict__ Wih_b,
    const float* __restrict__ bih_b, const float* __restrict__ bhh_b,
    const float* __restrict__ Wh2p, const float* __restrict__ bh2p,
    const float* __restrict__ embed_dec, const float* __restrict__ Wz2h,
    const float* __restrict__ bz2h, const float* __restrict__ Wih1,
    const float* __restrict__ Whh1, const float* __restrict__ bih1,
    const float* __restrict__ bhh1, const float* __restrict__ Wih2,
    const float* __restrict__ Whh2, const float* __restrict__ bih2,
    const float* __restrict__ bhh2, const float* __restrict__ Wh2o,
    const float* __restrict__ bh2o, float* __restrict__ out) {
  __shared__ SharedMem smem;
  const int tid  = threadIdx.x;
  const int lane = tid & 31;
  const int wv   = tid >> 5;
  const int b0   = blockIdx.x * ROWS_PER_WG + wv * ROWS_PER_WAVE;
  const int myM  = lane & 15;
  const int ch   = lane >> 4;
  WaveScratch& W = smem.ws[wv];

  // ---- phase 0: encoder weights + all biases into LDS ----
  load_emb(embed_enc, smem.wpool + EMB_E, false, tid);
  load_wB(Wih_f, 150, 50, 2, 10, smem.wpool + WIHF, tid);
  load_wB(Whh_f, 150, 50, 2, 10, smem.wpool + WHHF, tid);
  load_wB(Wih_b, 150, 50, 2, 10, smem.wpool + WIHB, tid);
  load_wB(Wh2p,  40,  50, 2, 3,  smem.wpool + WH2P, tid);
  load_wB(Wz2h,  50,  20, 1, 4,  smem.wpool + WZ2H, tid);
  load_bias(bih_f, 150, 160, smem.bias + BIHF, tid);
  load_bias(bhh_f, 150, 160, smem.bias + BHHF, tid);
  load_bias(bih_b, 150, 160, smem.bias + BIHB, tid);
  load_bias(bhh_b, 150, 160, smem.bias + BHHB, tid);
  load_bias(bh2p,  40,  48,  smem.bias + BH2P, tid);
  load_bias(bz2h,  50,  64,  smem.bias + BZ2H, tid);
  load_bias(bih1, 150, 160, smem.bias + BIH1, tid);
  load_bias(bhh1, 150, 160, smem.bias + BHH1, tid);
  load_bias(bih2, 150, 160, smem.bias + BIH2, tid);
  load_bias(bhh2, 150, 160, smem.bias + BHH2, tid);
  load_bias(bh2o, 128, 128, smem.bias + BH2O, tid);
  __syncthreads();

  // per-wave state init (also zeroes K-padding columns, which stay zero)
  for (int i = lane; i < 16 * 64; i += 32) { W.hbuf[i] = (_Float16)0.f; W.h2buf[i] = (_Float16)0.f; }
  for (int i = lane; i < 16 * 96; i += 32) { W.x1buf[i] = (_Float16)0.f; W.hz2buf[i] = (_Float16)0.f; }
  for (int i = lane; i < 16 * 32; i += 32) W.zbuf[i] = (_Float16)0.f;
  WAVE_SYNC();

  // ---- encoder: 64 forward GRU steps ----
  for (int t = 0; t < TSTEPS; ++t) {
    int tk = x[(size_t)(b0 + myM) * TSTEPS + t];
    if (t + 1 < TSTEPS)
      __builtin_prefetch(&x[(size_t)(b0 + myM) * TSTEPS + t + 1], 0, 3);
    wave_gemm<2, 10>(smem.wpool + EMB_E + tk * 64, smem.wpool + WIHF,
                     smem.bias + BIHF, W.gi, lane);
    wave_gemm<2, 10>(W.hbuf + myM * 64, smem.wpool + WHHF,
                     smem.bias + BHHF, W.gh, lane);
    WAVE_SYNC();
    gru_combine(W, W.hbuf, nullptr, lane);
    WAVE_SYNC();
  }

  // ---- backward cell on x[:, T-1] with h0 = 0 (gh == bhh_b) ----
  {
    int tk = x[(size_t)(b0 + myM) * TSTEPS + (TSTEPS - 1)];
    wave_gemm<2, 10>(smem.wpool + EMB_E + tk * 64, smem.wpool + WIHB,
                     smem.bias + BIHB, W.gi, lane);
    WAVE_SYNC();
    const float* bhhb = smem.bias + BHHB;
    for (int i = 0; i < 25; ++i) {
      int c = ch + 2 * i;
      float r  = fsig(W.gi[CM(c, myM)] + bhhb[c]);
      float zg = fsig(W.gi[CM(50 + c, myM)] + bhhb[50 + c]);
      float nn = ftanh(W.gi[CM(100 + c, myM)] + r * bhhb[100 + c]);
      float henc = (float)W.hbuf[myM * 64 + c] + (1.f - zg) * nn; // f_last + b_last
      W.hbuf[myM * 64 + c] = (_Float16)henc;
    }
    WAVE_SYNC();
  }

  // ---- latent: p = h_enc @ Wh2p.T + bh2p ; z = mu ----
  wave_gemm<2, 3>(W.hbuf + myM * 64, smem.wpool + WH2P, smem.bias + BH2P,
                  W.gi, lane);
  WAVE_SYNC();
  {
    const size_t muOff = (size_t)BATCH * MAXLEN * VOCAB;
    const size_t lvOff = muOff + (size_t)BATCH * LDIM;
    for (int i = lane; i < 16 * LDIM; i += 32) {
      int m = i / LDIM, l = i % LDIM;
      float mu = W.gi[CM(l, m)];
      float lv = W.gi[CM(LDIM + l, m)];
      out[muOff + (size_t)(b0 + m) * LDIM + l] = mu;
      out[lvOff + (size_t)(b0 + m) * LDIM + l] = lv;
      _Float16 zh = (_Float16)mu;
      W.zbuf[m * 32 + l] = zh;
      W.x1buf[m * 96 + 50 + l] = zh;
      W.hz2buf[m * 96 + 50 + l] = zh;
    }
    WAVE_SYNC();
  }

  // ---- h_init = z @ Wz2h.T + bz2h -> h1, h2 ----
  wave_gemm<1, 4>(W.zbuf + myM * 32, smem.wpool + WZ2H, smem.bias + BZ2H,
                  W.gh, lane);
  WAVE_SYNC();
  for (int i = 0; i < 25; ++i) {
    int c = ch + 2 * i;
    _Float16 h0 = (_Float16)W.gh[CM(c, myM)];
    W.hbuf[myM * 64 + c] = h0;
    W.h2buf[myM * 64 + c] = h0;
  }
  if (lane < 16) W.tok[lane] = SOS_TOK;
  WAVE_SYNC();

  // ---- phase switch: swap in decoder weights (SiLU folded into embed) ----
  __syncthreads();
  load_emb(embed_dec, smem.wpool + ETAB, true, tid);
  load_wB(Wih1, 150, 70, 3, 10, smem.wpool + WIH1, tid);
  load_wB(Whh1, 150, 50, 2, 10, smem.wpool + WHH1, tid);
  load_wB(Wih2, 150, 50, 2, 10, smem.wpool + WIH2, tid);
  load_wB(Whh2, 150, 50, 2, 10, smem.wpool + WHH2, tid);
  load_wB(Wh2o, 128, 70, 3, 8,  smem.wpool + WH2O, tid);
  __syncthreads();

  // ---- decoder: 64 steps with argmax feedback ----
  for (int t = 0; t < MAXLEN; ++t) {
    // x1[:, 0:50] = silu(embed_dec[tok]) ; copy 2 halves per DS op
    for (int m = 0; m < 16; ++m) {
      const unsigned int* er =
          (const unsigned int*)(smem.wpool + ETAB + W.tok[m] * 64);
      unsigned int* xr = (unsigned int*)(W.x1buf + m * 96);
      for (int c2 = lane; c2 < 25; c2 += 32) xr[c2] = er[c2];
    }
    WAVE_SYNC();
    // GRU 1
    wave_gemm<3, 10>(W.x1buf + myM * 96, smem.wpool + WIH1, smem.bias + BIH1,
                     W.gi, lane);
    wave_gemm<2, 10>(W.hbuf + myM * 64, smem.wpool + WHH1, smem.bias + BHH1,
                     W.gh, lane);
    WAVE_SYNC();
    gru_combine(W, W.hbuf, nullptr, lane);   // h1n
    WAVE_SYNC();
    // GRU 2 (input = h1n)
    wave_gemm<2, 10>(W.hbuf + myM * 64, smem.wpool + WIH2, smem.bias + BIH2,
                     W.gi, lane);
    wave_gemm<2, 10>(W.h2buf + myM * 64, smem.wpool + WHH2, smem.bias + BHH2,
                     W.gh, lane);
    WAVE_SYNC();
    gru_combine(W, W.h2buf, W.hz2buf, lane); // h2n, also into [h2|z] buffer
    WAVE_SYNC();
    // logits = [h2n | z] @ Wh2o.T + bh2o  (into gi scratch, column-major)
    wave_gemm<3, 8>(W.hz2buf + myM * 96, smem.wpool + WH2O, smem.bias + BH2O,
                    W.gi, lane);
    WAVE_SYNC();
    // log-softmax + argmax (one lane per batch row)
    if (lane < 16) {
      int m = lane;
      float mx = -1e30f; int am = 0;
      for (int c = 0; c < VOCAB; ++c) {
        float v = W.gi[CM(c, m)];
        if (v > mx) { mx = v; am = c; }
      }
      float s = 0.f;
      for (int c = 0; c < VOCAB; ++c) s += __expf(W.gi[CM(c, m)] - mx);
      W.logZ[m] = mx + __logf(s);
      W.tok[m] = am;
    }
    WAVE_SYNC();
    // write words[b][t][:] = logits - logZ  (coalesced along vocab dim)
    for (int i = lane; i < 16 * VOCAB; i += 32) {
      int m = i >> 7, c = i & 127;
      out[((size_t)(b0 + m) * MAXLEN + t) * VOCAB + c] =
          W.gi[CM(c, m)] - W.logZ[m];
    }
    WAVE_SYNC();
  }
}

extern "C" void kernel_launch(void* const* d_in, const int* in_sizes, int n_in,
                              void* d_out, int out_size, void* d_ws, size_t ws_size,
                              hipStream_t stream) {
  (void)in_sizes; (void)n_in; (void)d_ws; (void)ws_size; (void)out_size;
  const int*   x         = (const int*)d_in[0];
  const float* embed_enc = (const float*)d_in[1];
  const float* Wih_f     = (const float*)d_in[2];
  const float* Whh_f     = (const float*)d_in[3];
  const float* bih_f     = (const float*)d_in[4];
  const float* bhh_f     = (const float*)d_in[5];
  const float* Wih_b     = (const float*)d_in[6];
  // d_in[7] (Whh_b) is mathematically unused: backward cell sees h0 == 0.
  const float* bih_b     = (const float*)d_in[8];
  const float* bhh_b     = (const float*)d_in[9];
  const float* Wh2p      = (const float*)d_in[10];
  const float* bh2p      = (const float*)d_in[11];
  const float* embed_dec = (const float*)d_in[12];
  const float* Wz2h      = (const float*)d_in[13];
  const float* bz2h      = (const float*)d_in[14];
  const float* Wih1      = (const float*)d_in[15];
  const float* Whh1      = (const float*)d_in[16];
  const float* bih1      = (const float*)d_in[17];
  const float* bhh1      = (const float*)d_in[18];
  const float* Wih2      = (const float*)d_in[19];
  const float* Whh2      = (const float*)d_in[20];
  const float* bih2      = (const float*)d_in[21];
  const float* bhh2      = (const float*)d_in[22];
  const float* Wh2o      = (const float*)d_in[23];
  const float* bh2o      = (const float*)d_in[24];
  float* out = (float*)d_out;

  dim3 grid(BATCH / ROWS_PER_WG);   // 128 workgroups
  dim3 block(NWAVES * 32);          // 4 wave32s -> 64 batch rows per WG
  textvae_kernel<<<grid, block, 0, stream>>>(
      x, embed_enc, Wih_f, Whh_f, bih_f, bhh_f, Wih_b, bih_b, bhh_b,
      Wh2p, bh2p, embed_dec, Wz2h, bz2h, Wih1, Whh1, bih1, bhh1,
      Wih2, Whh2, bih2, bhh2, Wh2o, bh2o, out);
}
